// MAB_73108933313260
// MI455X (gfx1250) — matrix operand order repaired and checked by reading, not compiled, not run on hardware
//
#include <hip/hip_runtime.h>

#define NGRAPHS 16
#define MAXLEN  1024
#define DV      256
#define NHEADS  8

typedef __attribute__((ext_vector_type(16))) _Float16 v16h;
typedef __attribute__((ext_vector_type(8)))  _Float16 v8h;
typedef __attribute__((ext_vector_type(2)))  __fp16   v2h;   // cvt_pkrtz result type
typedef __attribute__((ext_vector_type(8)))  float    v8f;
typedef __attribute__((ext_vector_type(4)))  float    f32x4;
typedef __attribute__((ext_vector_type(4)))  unsigned int u32x4;
typedef __attribute__((ext_vector_type(8)))  int          i32x8;
typedef __attribute__((ext_vector_type(4)))  int          i32x4;

__device__ __forceinline__ v8f wmma16(v16h a, v16h b, v8f c) {
    return __builtin_amdgcn_wmma_f32_16x16x32_f16(
        false, a, false, b, (short)0, c, false, false);
}

__device__ __forceinline__ v16h join16(v8h lo, v8h hi) {
    v16h r;
    #pragma unroll
    for (int i = 0; i < 8; ++i) { r[i] = lo[i]; r[8 + i] = hi[i]; }
    return r;
}

__device__ __forceinline__ unsigned rfl(unsigned x) {
    return (unsigned)__builtin_amdgcn_readfirstlane((int)x);
}

// ---------------------------------------------------------------------------
// TDM: async 2D f16 tile (tile1 rows x tile0 elems) global -> LDS, row-major.
// D# per CDNA5 ISA ch.8: group0 = {count|flags, lds_addr, global_addr, type=2},
// group1 = {mask/data_size, dims, tile dims, strides}. Rows/cols >= tensor_dim
// read as zero (free ragged-tail masking). All dwords forced uniform.
// ---------------------------------------------------------------------------
__device__ __forceinline__ void tdm_load_2d_f16(unsigned ldsByteOff, const _Float16* g,
                                                unsigned dim0, unsigned dim1,
                                                unsigned strideElems,
                                                unsigned tile0, unsigned tile1) {
    unsigned long long ga = (unsigned long long)(uintptr_t)g;
    unsigned galo = rfl((unsigned)ga);
    unsigned gahi = rfl((unsigned)(ga >> 32));
    dim0 = rfl(dim0); dim1 = rfl(dim1); strideElems = rfl(strideElems);
    tile0 = rfl(tile0); tile1 = rfl(tile1); ldsByteOff = rfl(ldsByteOff);

    u32x4 g0;
    g0[0] = 1u;                                   // count=1 (valid user D#)
    g0[1] = ldsByteOff;                           // lds_addr (bytes)
    g0[2] = galo;                                 // global_addr[31:0]
    g0[3] = (gahi & 0x01ffffffu) | (2u << 30);    // global_addr[56:32] | type=2

    i32x8 g1;
    g1[0] = (int)(1u << 16);                                        // data_size=2B
    g1[1] = (int)((dim0 & 0xffffu) << 16);                          // tensor_dim0.lo
    g1[2] = (int)((dim0 >> 16) | ((dim1 & 0xffffu) << 16));         // dim0.hi|dim1.lo
    g1[3] = (int)((dim1 >> 16) | ((tile0 & 0xffffu) << 16));        // dim1.hi|tile0
    g1[4] = (int)(tile1 & 0xffffu);                                 // tile1|tile2=0
    g1[5] = (int)strideElems;                                       // dim0_stride lo
    g1[6] = 0;
    g1[7] = 0;

    i32x4 z4; z4[0] = z4[1] = z4[2] = z4[3] = 0;                    // 2D: dims 2+ unused
    i32x8 z8;
    #pragma unroll
    for (int i = 0; i < 8; ++i) z8[i] = 0;
    __builtin_amdgcn_tensor_load_to_lds(g0, g1, z4, z4, z8, 0);
}

// ---------------------------------------------------------------------------
__global__ void starts_kernel(const int* __restrict__ batch, int Ntot,
                              int* __restrict__ starts) {
    int b = threadIdx.x;
    if (b > NGRAPHS) return;
    if (b == NGRAPHS) { starts[NGRAPHS] = Ntot; return; }
    int lo = 0, hi = Ntot;
    while (lo < hi) {
        int mid = (lo + hi) >> 1;
        if (batch[mid] < b) lo = mid + 1; else hi = mid;
    }
    starts[b] = lo;
}

// ---------------------------------------------------------------------------
// Z = X[Mx256] @ W[256x256] + bias, with optional fused outputs:
//   outF (fp32, mode1: R + relu(Z))    outH (f16, *hscale, row-major)
//   outT (f16, transposed [col][row], row pitch = M)
// 256 thr = 8 waves; block tile 128x64; each wave 32x32 (4 WMMA / k-step).
// ---------------------------------------------------------------------------
__global__ void gemm_kernel(const float* __restrict__ X, const float* __restrict__ W,
                            const float* __restrict__ bias, const float* __restrict__ R,
                            float* __restrict__ outF,
                            _Float16* __restrict__ outH, float hscale,
                            _Float16* __restrict__ outT,
                            int M, int mode) {
    __shared__ __align__(16) _Float16 Xs[128][32];   // [row][k]
    __shared__ __align__(16) _Float16 Wt[64][32];    // [n][k] (transposed W tile)

    const int tid  = threadIdx.x;
    const int lane = tid & 31;
    const int wave = tid >> 5;
    const int wm   = wave >> 1;        // 0..3 -> 32-row subtile
    const int wn   = wave & 1;         // 0..1 -> 32-col subtile
    const int half = lane >> 4;
    const int lr   = lane & 15;
    const int m0   = blockIdx.x * 128;
    const int n0   = blockIdx.y * 64;

    v8f c00 = {}, c01 = {}, c10 = {}, c11 = {};

    for (int kk = 0; kk < 256; kk += 32) {
        __syncthreads();
        { // stage X: clamped base -> 4 unconditional b128 loads, branchless zeroing
            int row = tid >> 1;
            int kb  = (tid & 1) * 16;
            int gr  = m0 + row;
            bool ok = gr < M;
            const f32x4* s4 = (const f32x4*)(X + (size_t)(ok ? gr : 0) * 256 + kk + kb);
            f32x4 f[4];
            #pragma unroll
            for (int j = 0; j < 4; ++j) f[j] = s4[j];
            #pragma unroll
            for (int j = 0; j < 4; ++j) {
                #pragma unroll
                for (int e = 0; e < 4; ++e) f[j][e] = ok ? f[j][e] : 0.f;   // v_cndmask
            }
            v8h a0, a1;
            #pragma unroll
            for (int j = 0; j < 2; ++j) {
                v2h p0 = __builtin_amdgcn_cvt_pkrtz(f[j][0], f[j][1]);
                v2h p1 = __builtin_amdgcn_cvt_pkrtz(f[j][2], f[j][3]);
                a0[4 * j]     = (_Float16)p0[0]; a0[4 * j + 1] = (_Float16)p0[1];
                a0[4 * j + 2] = (_Float16)p1[0]; a0[4 * j + 3] = (_Float16)p1[1];
            }
            #pragma unroll
            for (int j = 0; j < 2; ++j) {
                v2h p0 = __builtin_amdgcn_cvt_pkrtz(f[2 + j][0], f[2 + j][1]);
                v2h p1 = __builtin_amdgcn_cvt_pkrtz(f[2 + j][2], f[2 + j][3]);
                a1[4 * j]     = (_Float16)p0[0]; a1[4 * j + 1] = (_Float16)p0[1];
                a1[4 * j + 2] = (_Float16)p1[0]; a1[4 * j + 3] = (_Float16)p1[1];
            }
            *(v8h*)&Xs[row][kb]     = a0;
            *(v8h*)&Xs[row][kb + 8] = a1;
        }
        { // stage W transposed: coalesced-across-lanes rows, one b128 store
            int n  = tid & 63;
            int kc = (tid >> 6) * 8;
            v8h w;
            #pragma unroll
            for (int j = 0; j < 4; ++j) {
                float x0 = W[(size_t)(kk + kc + 2 * j)     * 256 + n0 + n];
                float x1 = W[(size_t)(kk + kc + 2 * j + 1) * 256 + n0 + n];
                v2h p = __builtin_amdgcn_cvt_pkrtz(x0, x1);
                w[2 * j] = (_Float16)p[0]; w[2 * j + 1] = (_Float16)p[1];
            }
            *(v8h*)&Wt[n][kc] = w;
        }
        __syncthreads();

        const _Float16* xr0 = &Xs[wm * 32 + lr][0];
        const _Float16* xr1 = &Xs[wm * 32 + 16 + lr][0];
        v16h a0 = join16(*(const v8h*)(xr0 + half * 8), *(const v8h*)(xr0 + 16 + half * 8));
        v16h a1 = join16(*(const v8h*)(xr1 + half * 8), *(const v8h*)(xr1 + 16 + half * 8));

        const _Float16* wc0 = &Wt[wn * 32 + lr][0];
        const _Float16* wc1 = &Wt[wn * 32 + 16 + lr][0];
        v16h b0 = join16(*(const v8h*)(wc0 + half * 8), *(const v8h*)(wc0 + 16 + half * 8));
        v16h b1 = join16(*(const v8h*)(wc1 + half * 8), *(const v8h*)(wc1 + 16 + half * 8));

        c00 = wmma16(a0, b0, c00);
        c01 = wmma16(a0, b1, c01);
        c10 = wmma16(a1, b0, c10);
        c11 = wmma16(a1, b1, c11);
    }

    auto emit = [&](v8f c, int rbase, int cbase) {
        #pragma unroll
        for (int v = 0; v < 8; ++v) {
            int row = rbase + v + 8 * half;
            if (row >= M) continue;
            int col = cbase + lr;
            float z = c[v] + bias[col];
            if (mode == 1) z = R[(size_t)row * 256 + col] + fmaxf(z, 0.f);
            if (outF) outF[(size_t)row * 256 + col] = z;
            if (outH) outH[(size_t)row * 256 + col] = (_Float16)(z * hscale);
            if (outT) outT[(size_t)col * M + row] = (_Float16)z;
        }
    };
    emit(c00, m0 + wm * 32,      n0 + wn * 32);
    emit(c01, m0 + wm * 32,      n0 + wn * 32 + 16);
    emit(c10, m0 + wm * 32 + 16, n0 + wn * 32);
    emit(c11, m0 + wm * 32 + 16, n0 + wn * 32 + 16);
}

// ---------------------------------------------------------------------------
// Flash attention. grid (MAXLEN/64, H, B), 128 thr = 4 waves (16 q-rows each).
// K tile (32x32 f16, row-major) and V tile (32x32 f16, already transposed
// [d][key] thanks to vp_t layout) are DMA'd to LDS by the Tensor Data Mover.
// Dynamic LDS only, so D# lds_addr offsets are the literal byte offsets:
//   [0,2048)    Ks[key][d]
//   [2048,4096) Vt[d][key]
//   [4096,8192) P[wave][row][key]
// ---------------------------------------------------------------------------
__global__ void attn_kernel(const float* __restrict__ qp,
                            const _Float16* __restrict__ qp_h,
                            const _Float16* __restrict__ kp_h,
                            const _Float16* __restrict__ vp_t,
                            const int* __restrict__ starts,
                            float* __restrict__ attn, int Ntot) {
    const int b  = blockIdx.z;
    const int h  = blockIdx.y;
    const int s0 = starts[b];
    const int L  = starts[b + 1] - s0;
    const int qbase = blockIdx.x * 64;
    if (qbase >= L) return;            // whole block exits before any barrier

    extern __shared__ _Float16 smem[];
    _Float16* Ks = smem;               // 32x32 [key][d]
    _Float16* Vt = smem + 1024;        // 32x32 [d][key]

    const int tid  = threadIdx.x;
    const int lane = tid & 31;
    const int wave = tid >> 5;
    const int half = lane >> 4;
    const int lr   = lane & 15;
    const int q0   = qbase + wave * 16;
    _Float16* Pw = smem + 2048 + wave * 512;   // 16x32 per-wave P tile

    // Q fragment (qp_h is pre-scaled by 1/sqrt(DH))
    v16h qa;
    {
        int r = s0 + q0 + lr;
        if (r >= Ntot) r = Ntot - 1;
        const _Float16* src = qp_h + (size_t)r * DV + h * 32;
        qa = join16(*(const v8h*)(src + half * 8), *(const v8h*)(src + 16 + half * 8));
    }

    v8f o0 = {}, o1 = {};
    float mrun[8], lrun[8];
    #pragma unroll
    for (int v = 0; v < 8; ++v) { mrun[v] = -3.0e38f; lrun[v] = 0.f; }

    const int ktiles = (L + 31) >> 5;
    for (int jt = 0; jt < ktiles; ++jt) {
        const int k0 = jt * 32;

        __syncthreads();               // previous tile fully consumed
        if (wave == 0) {
            // K tile: 32 keys x 32 dims, row-major; OOB keys -> 0
            tdm_load_2d_f16(0,    kp_h + (size_t)(s0 + k0) * DV + h * 32,
                            32, (unsigned)(L - k0), DV, 32, 32);
            // V tile: rows = head dims, cols = keys; OOB keys -> 0
            tdm_load_2d_f16(2048, vp_t + (size_t)(h * 32) * Ntot + (s0 + k0),
                            (unsigned)(L - k0), 32, (unsigned)Ntot, 32, 32);
            __builtin_amdgcn_s_wait_tensorcnt(0);
        }
        __syncthreads();

        // S = Q K^T : B-fragment = contiguous Ks rows
        const _Float16* kr0 = Ks + (size_t)lr * 32;
        const _Float16* kr1 = Ks + (size_t)(16 + lr) * 32;
        v16h kb0 = join16(*(const v8h*)(kr0 + half * 8), *(const v8h*)(kr0 + 16 + half * 8));
        v16h kb1 = join16(*(const v8h*)(kr1 + half * 8), *(const v8h*)(kr1 + 16 + half * 8));
        v8f zc = {};
        v8f sf0 = wmma16(qa, kb0, zc);
        v8f sf1 = wmma16(qa, kb1, zc);

        const float NEG = -1.0e30f;
        const bool ok0 = (k0 + lr)      < L;
        const bool ok1 = (k0 + 16 + lr) < L;

        float e0v[8], e1v[8];
        #pragma unroll
        for (int v = 0; v < 8; ++v) {
            float s0v = ok0 ? sf0[v] : NEG;
            float s1v = ok1 ? sf1[v] : NEG;
            float mx = fmaxf(s0v, s1v);
            #pragma unroll
            for (int d = 1; d < 16; d <<= 1) mx = fmaxf(mx, __shfl_xor(mx, d, 32));
            float mnew  = fmaxf(mrun[v], mx);
            float scale = __expf(mrun[v] - mnew);
            float e0 = __expf(s0v - mnew);
            float e1 = __expf(s1v - mnew);
            float rs = e0 + e1;
            #pragma unroll
            for (int d = 1; d < 16; d <<= 1) rs += __shfl_xor(rs, d, 32);
            lrun[v] = lrun[v] * scale + rs;
            mrun[v] = mnew;
            o0[v] *= scale;
            o1[v] *= scale;
            e0v[v] = e0; e1v[v] = e1;
        }

        // C layout -> A layout via per-wave LDS round trip
        #pragma unroll
        for (int v = 0; v < 8; ++v) {
            int row = v + 8 * half;
            Pw[row * 32 + lr]      = (_Float16)e0v[v];
            Pw[row * 32 + 16 + lr] = (_Float16)e1v[v];
        }
        __syncthreads();

        const _Float16* pr = Pw + (size_t)lr * 32;
        v16h pa = join16(*(const v8h*)(pr + half * 8), *(const v8h*)(pr + 16 + half * 8));

        const _Float16* vr0 = Vt + (size_t)lr * 32;
        const _Float16* vr1 = Vt + (size_t)(16 + lr) * 32;
        v16h vb0 = join16(*(const v8h*)(vr0 + half * 8), *(const v8h*)(vr0 + 16 + half * 8));
        v16h vb1 = join16(*(const v8h*)(vr1 + half * 8), *(const v8h*)(vr1 + 16 + half * 8));

        o0 = wmma16(pa, vb0, o0);
        o1 = wmma16(pa, vb1, o1);
    }

    #pragma unroll
    for (int v = 0; v < 8; ++v) {
        int rowLocal = q0 + v + 8 * half;
        if (rowLocal >= L) continue;
        size_t base = (size_t)(s0 + rowLocal) * DV + h * 32;
        float inv = 1.f / lrun[v];
        attn[base + lr]      = o0[v] * inv + qp[base + lr];
        attn[base + 16 + lr] = o1[v] * inv + qp[base + 16 + lr];
    }
}

// ---------------------------------------------------------------------------
extern "C" void kernel_launch(void* const* d_in, const int* in_sizes, int n_in,
                              void* d_out, int out_size, void* d_ws, size_t ws_size,
                              hipStream_t stream) {
    const float* q       = (const float*)d_in[0];
    const float* k       = (const float*)d_in[1];
    const int*   batch_q = (const int*)d_in[2];
    const float* Wq = (const float*)d_in[4];
    const float* bq = (const float*)d_in[5];
    const float* Wk = (const float*)d_in[6];
    const float* bk = (const float*)d_in[7];
    const float* Wv = (const float*)d_in[8];
    const float* bv = (const float*)d_in[9];
    const float* Wo = (const float*)d_in[10];
    const float* bo = (const float*)d_in[11];

    const int N = in_sizes[0] / 256;

    float*    qp    = (float*)d_ws;                          // N*256 f32
    float*    attnb = qp + (size_t)N * 256;                  // N*256 f32
    _Float16* qp_h  = (_Float16*)(attnb + (size_t)N * 256);  // N*256 f16 (pre-scaled)
    _Float16* kp_h  = qp_h + (size_t)N * 256;                // N*256 f16
    _Float16* vp_t  = kp_h + (size_t)N * 256;                // 256*N f16 (transposed)
    int*      starts = (int*)(vp_t + (size_t)N * 256);       // B+1

    starts_kernel<<<1, 32, 0, stream>>>(batch_q, N, starts);

    dim3 gg((N + 127) / 128, 4);
    const float qscale = 0.17677669529663688f;   // 1/sqrt(32)
    gemm_kernel<<<gg, 256, 0, stream>>>(q, Wq, bq, nullptr, qp, qp_h, qscale, nullptr, N, 0);
    gemm_kernel<<<gg, 256, 0, stream>>>(k, Wk, bk, nullptr, nullptr, kp_h, 1.f, nullptr, N, 0);
    gemm_kernel<<<gg, 256, 0, stream>>>(k, Wv, bv, nullptr, nullptr, nullptr, 1.f, vp_t, N, 0);

    dim3 ga(MAXLEN / 64, NHEADS, NGRAPHS);
    attn_kernel<<<ga, 128, 8192, stream>>>(qp, qp_h, kp_h, vp_t, starts, attnb, N);

    gemm_kernel<<<gg, 256, 0, stream>>>(attnb, Wo, bo, attnb, (float*)d_out, nullptr, 1.f, nullptr, N, 1);
}
